// HierarchicalEngramMemory_18992345383229
// MI455X (gfx1250) — matrix operand order repaired and checked by reading, not compiled, not run on hardware
//
#include <hip/hip_runtime.h>

// ---------------------------------------------------------------------------
// Hierarchical engram retrieval for MI455X (gfx1250, wave32, WMMA).
// Memory-bound on the 1.07 GB L3 SDR bank read (~46 us at 23.3 TB/s).
// Similarities via v_wmma_f32_16x16x32_bf16 (exact for 0/1 data); B-operand
// data is software-pipelined (load k+1 while converting/WMMA-ing k) so the
// global_load_b128s clause up instead of stalling on s_wait_loadcnt 0.
// Threshold (overlap >= 12) is statistically sparse -> atomic candidate
// append + tiny deterministic top-16 sort/gather pass.
// ---------------------------------------------------------------------------

typedef __attribute__((ext_vector_type(16))) __bf16 v16bf;
typedef __attribute__((ext_vector_type(8)))  __bf16 v8bf;
typedef __attribute__((ext_vector_type(8)))  float  v8f;
typedef __attribute__((ext_vector_type(4)))  float  v4f;

#define SDR_DIM      2048
#define CONTENT_DIM  384
#define NQ           16          // batch of queries
#define CAP          256         // candidate capacity per (tier,query)
#define THRESH_CNT   11.5f       // overlap >= 12  <=>  sim >= 0.3
#define QSTRIDE      2056        // bf16 elems per LDS query row (+8 pad -> 4112B, conflict free)
#define WAVES        8
#define TPW          4           // 16-row tiles per wave
#define TILE_STRIDE  (16 * SDR_DIM)         // floats between tile bases (128KB, fits imm offset)
#define ROWS_PER_BLOCK (16 * WAVES * TPW)   // 512

__global__ void engram_zero_kernel(int* __restrict__ counters) {
  if (threadIdx.x < 48) counters[threadIdx.x] = 0;
}

__global__ __launch_bounds__(256)
void engram_sim_kernel(const float* __restrict__ query,           // [16][2048]
                       const float* __restrict__ bank,            // [nrows][2048]
                       const unsigned char* __restrict__ valid,   // [nrows]
                       int tier,
                       int*   __restrict__ counters,              // [48]
                       float* __restrict__ candSim,               // [48][CAP]
                       int*   __restrict__ candIdx)               // [48][CAP]
{
  extern __shared__ char smem_raw[];
  __bf16* qlds = (__bf16*)smem_raw;

  const int tid = threadIdx.x;

  // Stage the 16x2048 query into LDS as bf16 (exact for 0/1 values).
  for (int i = tid; i < NQ * SDR_DIM; i += 256) {
    const int m = i >> 11;
    const int k = i & (SDR_DIM - 1);
    qlds[m * QSTRIDE + k] = (__bf16)query[i];
  }
  __syncthreads();

  const int wave  = tid >> 5;
  const int lane  = tid & 31;
  const int n     = lane & 15;        // bank row within tile / B column
  const int hi    = lane >> 4;        // lane group 0|1
  const int khalf = hi << 4;          // B frag: hi lanes carry K+16..K+31
  const int base8 = hi << 4;          // A frag: hi lanes carry K+16..K+31
  const __bf16* __restrict__ qrow = qlds + n * QSTRIDE;

  const int rbase = blockIdx.x * ROWS_PER_BLOCK + wave * (16 * TPW);

  // One per-lane base address; the 4 tiles sit at immediate offsets
  // t*TILE_STRIDE (128KB apart -> within signed 24-bit IOFFSET).
  const float* __restrict__ src = bank + (size_t)(rbase + n) * SDR_DIM + khalf;

  v8f accs[TPW] = {};

  // ---- software pipeline prologue: raw f32 chunk k0=0 + A frag k0=0 ----
  v4f raw[TPW * 4];
#pragma unroll
  for (int t = 0; t < TPW; ++t)
#pragma unroll
    for (int j = 0; j < 4; ++j)
      raw[t * 4 + j] = *(const v4f*)(src + t * TILE_STRIDE + j * 4);

  v8bf alo = *(const v8bf*)(qrow + base8);
  v8bf ahi = *(const v8bf*)(qrow + 8 + base8);

#pragma unroll 2
  for (int k0 = 0; k0 < SDR_DIM; k0 += 32) {
    const v16bf a = __builtin_shufflevector(alo, ahi,
        0, 1, 2, 3, 4, 5, 6, 7, 8, 9, 10, 11, 12, 13, 14, 15);

    // Issue next chunk's loads now (no consumer this iteration -> the
    // scheduler can clause them and overlap with converts + WMMAs below).
    v4f nxt[TPW * 4];
    v8bf nlo = alo, nhi = ahi;
    if (k0 + 32 < SDR_DIM) {
      const float* __restrict__ s2 = src + k0 + 32;
#pragma unroll
      for (int t = 0; t < TPW; ++t)
#pragma unroll
        for (int j = 0; j < 4; ++j)
          nxt[t * 4 + j] = *(const v4f*)(s2 + t * TILE_STRIDE + j * 4);
      nlo = *(const v8bf*)(qrow + (k0 + 32) + base8);
      nhi = *(const v8bf*)(qrow + (k0 + 32) + 8 + base8);
    } else {
#pragma unroll
      for (int i = 0; i < TPW * 4; ++i) nxt[i] = raw[i];
    }

    // Convert current chunk to bf16 and accumulate (exact: values are 0/1).
#pragma unroll
    for (int t = 0; t < TPW; ++t) {
      v16bf b;
#pragma unroll
      for (int j = 0; j < 4; ++j) {
        const v4f f = raw[t * 4 + j];
        b[j * 4 + 0] = (__bf16)f[0];
        b[j * 4 + 1] = (__bf16)f[1];
        b[j * 4 + 2] = (__bf16)f[2];
        b[j * 4 + 3] = (__bf16)f[3];
      }
      accs[t] = __builtin_amdgcn_wmma_f32_16x16x32_bf16(
          false, a, false, b, (short)0, accs[t], false, false);
    }

#pragma unroll
    for (int i = 0; i < TPW * 4; ++i) raw[i] = nxt[i];
    alo = nlo;
    ahi = nhi;
  }

  // Sparse candidate emission (divergence is fine here: WMMAs are done).
  const int qbase = tier * NQ;
#pragma unroll
  for (int t = 0; t < TPW; ++t) {
    const int row = rbase + t * 16 + n;
    const bool ok = valid[row] != 0;
#pragma unroll
    for (int e = 0; e < 8; ++e) {
      const float cntv = accs[t][e];          // exact integer overlap count
      if (ok && cntv >= THRESH_CNT) {
        const int m = e + (hi << 3);          // query index (C-layout M)
        const int slot = atomicAdd(&counters[qbase + m], 1);
        if (slot < CAP) {
          candSim[(qbase + m) * CAP + slot] = cntv / 40.0f;
          candIdx[(qbase + m) * CAP + slot] = row;
        }
      }
    }
  }
}

__global__ __launch_bounds__(256)
void engram_select_kernel(const int*   __restrict__ counters,
                          const float* __restrict__ candSim,
                          const int*   __restrict__ candIdx,
                          const float* __restrict__ c1,
                          const float* __restrict__ c2,
                          const float* __restrict__ c3,
                          float* __restrict__ out)   // [16][48][385]
{
  const int bq   = blockIdx.x & 15;
  const int tier = blockIdx.x >> 4;
  const float* __restrict__ content = (tier == 0) ? c1 : (tier == 1) ? c2 : c3;
  const int q = tier * NQ + bq;

  int cnt = counters[q];
  if (cnt > CAP) cnt = CAP;

  __shared__ float ssim[CAP];
  __shared__ int   sidx[CAP];
  __shared__ float topS[16];
  __shared__ int   topI[16];
  __shared__ int   nselS;

  for (int i = threadIdx.x; i < cnt; i += blockDim.x) {
    ssim[i] = candSim[q * CAP + i];
    sidx[i] = candIdx[q * CAP + i];
  }
  __syncthreads();

  // Deterministic top-16 by (sim desc, idx asc); cnt is tiny in practice.
  if (threadIdx.x == 0) {
    int ns = 0;
    for (int j = 0; j < 16 && j < cnt; ++j) {
      float bs = -1.0f; int bi = 0x7fffffff; int best = -1;
      for (int i = 0; i < cnt; ++i) {
        const float s = ssim[i];
        if (s > bs || (s == bs && sidx[i] < bi)) { bs = s; bi = sidx[i]; best = i; }
      }
      if (best < 0 || bs < 0.0f) break;
      topS[ns] = bs; topI[ns] = bi; ++ns;
      ssim[best] = -2.0f;
    }
    nselS = ns;
  }
  __syncthreads();
  const int nsel = nselS;

  // Write this (query, tier) slab fully: 16 slots x (384 content + 1 sim).
  const size_t obase = ((size_t)bq * 48 + tier * 16) * (CONTENT_DIM + 1);
  for (int e = threadIdx.x; e < 16 * (CONTENT_DIM + 1); e += blockDim.x) {
    const int j = e / (CONTENT_DIM + 1);
    const int c = e - j * (CONTENT_DIM + 1);
    float v = 0.0f;
    if (j < nsel) {
      v = (c == CONTENT_DIM) ? topS[j]
                             : content[(size_t)topI[j] * CONTENT_DIM + c];
    }
    out[obase + (size_t)j * (CONTENT_DIM + 1) + c] = v;
  }
}

extern "C" void kernel_launch(void* const* d_in, const int* in_sizes, int n_in,
                              void* d_out, int out_size, void* d_ws, size_t ws_size,
                              hipStream_t stream) {
  (void)in_sizes; (void)n_in; (void)out_size; (void)ws_size;

  const float*         query      = (const float*)d_in[0];
  const float*         l1_sdr     = (const float*)d_in[1];
  const float*         l1_content = (const float*)d_in[2];
  const unsigned char* l1_valid   = (const unsigned char*)d_in[3];
  const float*         l2_sdr     = (const float*)d_in[4];
  const float*         l2_content = (const float*)d_in[5];
  const unsigned char* l2_valid   = (const unsigned char*)d_in[6];
  const float*         l3_sdr     = (const float*)d_in[7];
  const float*         l3_content = (const float*)d_in[8];
  const unsigned char* l3_valid   = (const unsigned char*)d_in[9];
  // d_in[10] = top_k (fixed at 16 by the reference's output shape)

  int*   counters = (int*)d_ws;
  float* candSim  = (float*)((char*)d_ws + 256);
  int*   candIdx  = (int*)((char*)d_ws + 256 + 48 * CAP * sizeof(float));

  engram_zero_kernel<<<1, 64, 0, stream>>>(counters);

  const size_t ldsBytes = (size_t)NQ * QSTRIDE * 2;   // 65792 B query tile
  engram_sim_kernel<<<512    / ROWS_PER_BLOCK, 256, ldsBytes, stream>>>(
      query, l1_sdr, l1_valid, 0, counters, candSim, candIdx);
  engram_sim_kernel<<<8192   / ROWS_PER_BLOCK, 256, ldsBytes, stream>>>(
      query, l2_sdr, l2_valid, 1, counters, candSim, candIdx);
  engram_sim_kernel<<<131072 / ROWS_PER_BLOCK, 256, ldsBytes, stream>>>(
      query, l3_sdr, l3_valid, 2, counters, candSim, candIdx);

  engram_select_kernel<<<48, 256, 0, stream>>>(
      counters, candSim, candIdx, l1_content, l2_content, l3_content,
      (float*)d_out);
}
